// MixtureOfSoftmaxes_57724360458761
// MI455X (gfx1250) — matrix operand order repaired and checked by reading, not compiled, not run on hardware
//
#include <hip/hip_runtime.h>

// ---------------------------------------------------------------------------
// MixtureOfSoftmaxes for MI455X (gfx1250, wave32, WMMA)
// B=2, S=512 -> R=1024 rows; H=1024; D=H/2=512; K=4 experts; V=32000
//
// Roofline: expert GEMM = 268 GFLOP vs 524 MB one-shot read of expert_w
// (~22.5 us @ 23.3 TB/s) -> AI ~512 FLOP/B, compute-bound: bf16 WMMA with
// f32 accumulate. Grid is ordered M-fastest so consecutive workgroups share
// the same B slab (perfect L2 temporal reuse -> W leaves HBM once).
// Per-expert logits (131 MB) are L2-resident (192 MB L2).
// ---------------------------------------------------------------------------

typedef __attribute__((ext_vector_type(16))) __bf16 v16bf;
typedef __attribute__((ext_vector_type(8)))  __bf16 v8bf;
typedef __attribute__((ext_vector_type(8)))  float  v8f;

#define ROWS 1024   // B*S
#define HDIM 1024
#define DDIM 512
#define NEXP 4
#define VDIM 32000

#define BM 128
#define BN 256
#define BK 32
#define LSTRIDE 40   // bf16 elems per LDS row (80 B -> keeps 16B alignment)

static __device__ __forceinline__ __bf16 f2bf(float f) { return (__bf16)f; }

// ---------------------------------------------------------------------------
// GEMM: C[M x N] (f32, row stride ldc) = A[M x K] (bf16, lda=K) * B[K x N]
// (fp32 in memory, row stride ldb, converted to bf16 while staging).
// Grid: (M/BM, N/BN)  -- M on x (fastest) so same-B workgroups are adjacent.
// 256 threads = 8 waves (wave grid 4M x 2N). Double-buffered LDS, one
// barrier per K-step, 16 WMMA per wave per K-step.
// ---------------------------------------------------------------------------
__global__ __launch_bounds__(256)
void gemm_bf16_wmma(const __bf16* __restrict__ A,
                    const float*  __restrict__ Bsrc,
                    float*        __restrict__ C,
                    int Kdim, int ldb, int ldc)
{
    __shared__ __align__(16) __bf16 As[2][BM][LSTRIDE];   // 2*128*40*2 = 20 KB
    __shared__ __align__(16) __bf16 Bs[2][BN][LSTRIDE];   // 2*256*40*2 = 40 KB

    const int tid  = threadIdx.x;
    const int lane = tid & 31;
    const int wave = tid >> 5;
    const int block_m = blockIdx.x * BM;   // M fastest -> B-slab L2 reuse
    const int block_n = blockIdx.y * BN;

    const int wm = (wave & 3) * 32;        // wave M offset: 0/32/64/96 (2 subtiles)
    const int wn = (wave >> 2) * 128;      // wave N offset: 0/128 (8 subtiles)

    // CDNA5 WMMA lane mapping
    const int khalf    = lane >> 4;        // 0: lanes 0-15, 1: lanes 16-31
    const int frag_row = lane & 15;

    v8f acc[2][8] = {};                    // 2 M-subtiles x 8 N-subtiles

    // A staging: thread -> 16 contiguous bf16 (32 B)
    const int a_row = tid >> 1;            // 0..127
    const int a_col = (tid & 1) * 16;      // 0 or 16

    // B staging: thread owns a column-pair (float2), 16 k's each
    const int b_pair = (tid & 127) * 2;    // column pair base: 0,2,..,254
    const int b_kh   = (tid >> 7) * 16;    // k range [b_kh, b_kh+16)

    const __bf16* aptr = A + (size_t)(block_m + a_row) * Kdim + a_col;
    const float*  bptr = Bsrc + (size_t)b_kh * ldb + block_n + b_pair;

    v8bf   areg0, areg1;
    float2 breg[16];

    // ---- prologue: stage tile 0 ----
    {
        areg0 = *(const v8bf*)(aptr);
        areg1 = *(const v8bf*)(aptr + 8);
        #pragma unroll
        for (int k = 0; k < 16; ++k) breg[k] = *(const float2*)(bptr + (size_t)k * ldb);
        *(v8bf*)&As[0][a_row][a_col]     = areg0;
        *(v8bf*)&As[0][a_row][a_col + 8] = areg1;
        #pragma unroll
        for (int c = 0; c < 2; ++c) {      // two halves of the 16 k's
            v8bf q0, q1;
            #pragma unroll
            for (int j = 0; j < 8; ++j) {
                q0[j] = f2bf(breg[c * 8 + j].x);
                q1[j] = f2bf(breg[c * 8 + j].y);
            }
            *(v8bf*)&Bs[0][b_pair][b_kh + c * 8]     = q0;
            *(v8bf*)&Bs[0][b_pair + 1][b_kh + c * 8] = q1;
        }
    }
    __syncthreads();

    int buf = 0;
    for (int k0 = 0; k0 < Kdim; k0 += BK) {
        const int nk = k0 + BK;
        const bool more = nk < Kdim;

        // ---- issue global loads for next tile (overlap with WMMA) ----
        if (more) {
            areg0 = *(const v8bf*)(aptr + nk);
            areg1 = *(const v8bf*)(aptr + nk + 8);
            const float* bp = bptr + (size_t)nk * ldb;
            #pragma unroll
            for (int k = 0; k < 16; ++k) breg[k] = *(const float2*)(bp + (size_t)k * ldb);
            if (nk + BK < Kdim)
                __builtin_prefetch(bptr + (size_t)(nk + BK) * ldb, 0, 1);
        }

        // ---- compute from LDS[buf] ----
        v16bf afrag[2];
        #pragma unroll
        for (int i = 0; i < 2; ++i) {
            const v8bf lo = *(const v8bf*)&As[buf][wm + 16 * i + frag_row][khalf * 8];
            const v8bf hi = *(const v8bf*)&As[buf][wm + 16 * i + frag_row][16 + khalf * 8];
            #pragma unroll
            for (int e = 0; e < 8; ++e) { afrag[i][e] = lo[e]; afrag[i][8 + e] = hi[e]; }
        }
        #pragma unroll
        for (int j = 0; j < 8; ++j) {
            v16bf bfrag;
            const v8bf lo = *(const v8bf*)&Bs[buf][wn + 16 * j + frag_row][khalf * 8];
            const v8bf hi = *(const v8bf*)&Bs[buf][wn + 16 * j + frag_row][16 + khalf * 8];
            #pragma unroll
            for (int e = 0; e < 8; ++e) { bfrag[e] = lo[e]; bfrag[8 + e] = hi[e]; }
            #pragma unroll
            for (int i = 0; i < 2; ++i)
                acc[i][j] = __builtin_amdgcn_wmma_f32_16x16x32_bf16(
                    false, afrag[i], false, bfrag, (short)0, acc[i][j], false, false);
        }

        // ---- store next tile into LDS[buf^1] ----
        if (more) {
            *(v8bf*)&As[buf ^ 1][a_row][a_col]     = areg0;
            *(v8bf*)&As[buf ^ 1][a_row][a_col + 8] = areg1;
            #pragma unroll
            for (int c = 0; c < 2; ++c) {
                v8bf q0, q1;
                #pragma unroll
                for (int j = 0; j < 8; ++j) {
                    q0[j] = f2bf(breg[c * 8 + j].x);
                    q1[j] = f2bf(breg[c * 8 + j].y);
                }
                *(v8bf*)&Bs[buf ^ 1][b_pair][b_kh + c * 8]     = q0;
                *(v8bf*)&Bs[buf ^ 1][b_pair + 1][b_kh + c * 8] = q1;
            }
        }
        __syncthreads();
        buf ^= 1;
    }

    // ---- epilogue: D VGPR r -> row wm+16i+r+8*khalf, col wn+16j+frag_row ----
    #pragma unroll
    for (int i = 0; i < 2; ++i) {
        #pragma unroll
        for (int j = 0; j < 8; ++j) {
            const int col = block_n + wn + 16 * j + frag_row;
            #pragma unroll
            for (int r = 0; r < 8; ++r) {
                const int row = block_m + wm + 16 * i + r + 8 * khalf;
                C[(size_t)row * ldc + col] = acc[i][j][r];
            }
        }
    }
}

// ---------------------------------------------------------------------------
// RMSNorm: one block per row, 256 threads x 4 elements -> bf16.
// ---------------------------------------------------------------------------
__global__ __launch_bounds__(256)
void rmsnorm_bf16(const float* __restrict__ x, const float* __restrict__ scale,
                  __bf16* __restrict__ y)
{
    __shared__ float red[256];
    const int row = blockIdx.x;
    const int tid = threadIdx.x;
    const float4 v = *(const float4*)(x + (size_t)row * HDIM + tid * 4);
    red[tid] = v.x * v.x + v.y * v.y + v.z * v.z + v.w * v.w;
    __syncthreads();
    #pragma unroll
    for (int s = 128; s > 0; s >>= 1) {
        if (tid < s) red[tid] += red[tid + s];
        __syncthreads();
    }
    const float inv = rsqrtf(red[0] * (1.0f / HDIM) + 1e-5f);
    const float4 sc = *(const float4*)(scale + tid * 4);
    __bf16* o = y + (size_t)row * HDIM + tid * 4;
    o[0] = f2bf(v.x * inv * sc.x);
    o[1] = f2bf(v.y * inv * sc.y);
    o[2] = f2bf(v.z * inv * sc.z);
    o[3] = f2bf(v.w * inv * sc.w);
}

// ---------------------------------------------------------------------------
// Gate finish: silu(g_pre + b_d) @ gate_up_w + b_u -> softmax over 4 experts.
// ---------------------------------------------------------------------------
__global__ __launch_bounds__(128)
void gate_finish(const float* __restrict__ g_pre,   // R x D
                 const float* __restrict__ bias_d,  // D
                 const float* __restrict__ w_up,    // D x 4
                 const float* __restrict__ bias_u,  // 4
                 float* __restrict__ gate_w)        // R x 4
{
    __shared__ float red[128][NEXP];
    const int row = blockIdx.x;
    const int tid = threadIdx.x;
    float p[NEXP] = {0.f, 0.f, 0.f, 0.f};
    for (int i = tid; i < DDIM; i += 128) {
        const float z = g_pre[(size_t)row * DDIM + i] + bias_d[i];
        const float s = z / (1.0f + __expf(-z));      // silu
        #pragma unroll
        for (int e = 0; e < NEXP; ++e) p[e] += s * w_up[i * NEXP + e];
    }
    #pragma unroll
    for (int e = 0; e < NEXP; ++e) red[tid][e] = p[e];
    __syncthreads();
    #pragma unroll
    for (int s = 64; s > 0; s >>= 1) {
        if (tid < s) {
            #pragma unroll
            for (int e = 0; e < NEXP; ++e) red[tid][e] += red[tid + s][e];
        }
        __syncthreads();
    }
    if (tid == 0) {
        float lg[NEXP], m = -1e30f, z = 0.f;
        #pragma unroll
        for (int e = 0; e < NEXP; ++e) { lg[e] = red[0][e] + bias_u[e]; m = fmaxf(m, lg[e]); }
        #pragma unroll
        for (int e = 0; e < NEXP; ++e) { lg[e] = __expf(lg[e] - m); z += lg[e]; }
        const float iz = 1.0f / z;
        #pragma unroll
        for (int e = 0; e < NEXP; ++e) gate_w[row * NEXP + e] = lg[e] * iz;
    }
}

// ---------------------------------------------------------------------------
// Row softmax over V per expert + weighted accumulate into out.
// Logits (131 MB) are L2-resident, so the 3-pass re-reads hit L2, not HBM.
// ---------------------------------------------------------------------------
__global__ __launch_bounds__(256)
void softmax_mix(const float* __restrict__ logits,  // R x V
                 const float* __restrict__ gate_w,  // R x 4
                 float* __restrict__ out,           // R x V
                 int k, int first, int last)
{
    __shared__ float red[256];
    const int row = blockIdx.x;
    const int tid = threadIdx.x;
    const float* lrow = logits + (size_t)row * VDIM;
    float* orow = out + (size_t)row * VDIM;

    float m = -1e30f;
    for (int v = tid; v < VDIM; v += 256) m = fmaxf(m, lrow[v]);
    red[tid] = m;
    __syncthreads();
    #pragma unroll
    for (int s = 128; s > 0; s >>= 1) {
        if (tid < s) red[tid] = fmaxf(red[tid], red[tid + s]);
        __syncthreads();
    }
    m = red[0];
    __syncthreads();

    float sum = 0.f;
    for (int v = tid; v < VDIM; v += 256) sum += __expf(lrow[v] - m);
    red[tid] = sum;
    __syncthreads();
    #pragma unroll
    for (int s = 128; s > 0; s >>= 1) {
        if (tid < s) red[tid] += red[tid + s];
        __syncthreads();
    }
    const float g = gate_w[row * NEXP + k];
    const float sc = g / red[0];

    for (int v = tid; v < VDIM; v += 256) {
        const float p = __expf(lrow[v] - m) * sc;
        float a = first ? p : (orow[v] + p);
        orow[v] = last ? __logf(a + 1e-10f) : a;
    }
}

// ---------------------------------------------------------------------------
extern "C" void kernel_launch(void* const* d_in, const int* in_sizes, int n_in,
                              void* d_out, int out_size, void* d_ws, size_t ws_size,
                              hipStream_t stream)
{
    const float* hidden      = (const float*)d_in[0];  // (B,S,H)
    const float* rms_scale   = (const float*)d_in[1];  // (H,)
    const float* gate_down_w = (const float*)d_in[2];  // (H, D)
    const float* gate_down_b = (const float*)d_in[3];  // (D,)
    const float* gate_up_w   = (const float*)d_in[4];  // (D, 4)
    const float* gate_up_b   = (const float*)d_in[5];  // (4,)
    const float* expert_w    = (const float*)d_in[6];  // (H, 4*V)
    float* out = (float*)d_out;                        // (B,S,V) fp32

    // workspace layout (needs ~139 MB)
    char* ws = (char*)d_ws;
    __bf16* x_bf   = (__bf16*)(ws);                        // 2 MB
    float*  g_pre  = (float*)(ws + (size_t)(2 << 20));     // 2 MB
    float*  gate_w = (float*)(ws + (size_t)(4 << 20));     // 16 KB
    float*  logits = (float*)(ws + (size_t)(8 << 20));     // 131 MB

    // 1) RMSNorm -> bf16
    rmsnorm_bf16<<<ROWS, 256, 0, stream>>>(hidden, rms_scale, x_bf);

    // 2) gate hidden: (1024x1024) @ (1024x512)   grid: (M, N) with M fastest
    gemm_bf16_wmma<<<dim3(ROWS / BM, DDIM / BN), 256, 0, stream>>>(
        x_bf, gate_down_w, g_pre, HDIM, DDIM, DDIM);

    // 3) silu + 512->4 + softmax4
    gate_finish<<<ROWS, 128, 0, stream>>>(g_pre, gate_down_b, gate_up_w,
                                          gate_up_b, gate_w);

    // 4) experts: GEMM + softmax/mix (stream order serializes logits reuse)
    for (int k = 0; k < NEXP; ++k) {
        gemm_bf16_wmma<<<dim3(ROWS / BM, VDIM / BN), 256, 0, stream>>>(
            x_bf, expert_w + (size_t)k * VDIM, logits,
            HDIM, NEXP * VDIM, VDIM);
        softmax_mix<<<ROWS, 256, 0, stream>>>(logits, gate_w, out,
                                              k, k == 0, k == NEXP - 1);
    }
}
